// VAE_64561948393707
// MI455X (gfx1250) — matrix operand order repaired and checked by reading, not compiled, not run on hardware
//
#include <hip/hip_runtime.h>
#include <hip/hip_bf16.h>
#include <math.h>
#include <stdint.h>

typedef __attribute__((ext_vector_type(16))) _Float16 v16h;
typedef __attribute__((ext_vector_type(8)))  _Float16 v8h;
typedef __attribute__((ext_vector_type(8)))  float    v8f;
typedef _Float16 h16;

#define DEV __device__ __forceinline__

// ---------------------------------------------------------------------------
// WMMA fragment helpers (layouts per CDNA5 ISA, wave32).
// All addresses are 16B-aligned by construction -> explicit v8h (b128) loads.
// A 16x32 f16: lanes 0-15 row M=lane, K={0..7,16..23}; lanes 16-31 K={8..15,24..31}
// ---------------------------------------------------------------------------
DEV v16h load_a_frag(const h16* X, int ldx, int row0, int k0, int rowclamp) {
  int l = threadIdx.x & 31;
  int hf = l >> 4, m = l & 15;
  int r = row0 + m; if (r > rowclamp) r = rowclamp;
  const v8h* p = (const v8h*)(X + (size_t)r * ldx + k0) + hf;
  v8h lo = p[0];
  v8h hi = p[2];
  v16h a;
#pragma unroll
  for (int i = 0; i < 8; ++i) { a[i] = lo[i]; a[i + 8] = hi[i]; }
  return a;
}

// Clamp-free variant for GEMMs whose M is an exact multiple of 16.
DEV v16h load_a_frag_nc(const h16* X, int ldx, int row0, int k0) {
  int l = threadIdx.x & 31;
  int hf = l >> 4, m = l & 15;
  const v8h* p = (const v8h*)(X + (size_t)(row0 + m) * ldx + k0) + hf;
  v8h lo = p[0];
  v8h hi = p[2];
  v16h a;
#pragma unroll
  for (int i = 0; i < 8; ++i) { a[i] = lo[i]; a[i + 8] = hi[i]; }
  return a;
}

// B 32x16 f16 from WT stored [N,K] row-major (contiguous K per output column):
// lanes 0-15 col N=lane, K={0..7,16..23}; lanes 16-31 K={8..15,24..31}
DEV v16h load_b_frag(const h16* WT, int ldk, int col0, int k0) {
  int l = threadIdx.x & 31;
  int hf = l >> 4, n = l & 15;
  const v8h* p = (const v8h*)(WT + (size_t)(col0 + n) * ldk + k0) + hf;
  v8h b0 = p[0], b1 = p[2];
  v16h b;
#pragma unroll
  for (int i = 0; i < 8; ++i) { b[i] = b0[i]; b[i + 8] = b1[i]; }
  return b;
}

DEV float elu1(float x) { return x > 0.f ? x : (__expf(x) - 1.f); }

DEV float pe_val(int pos, int d, int Dd) {
  float f = __expf(-(float)(d & ~1) * (9.210340371976184f / (float)Dd));
  float a = (float)pos * f;
  return (d & 1) ? __cosf(a) : __sinf(a);
}

// CDNA5 async global->LDS copy, 16 bytes per lane (GLOBAL_LOAD_ASYNC_TO_LDS_B128,
// GV mode, tracked by ASYNCcnt). LDS operand is the 32-bit LDS byte offset.
DEV void async_copy16(h16* lds_dst, const h16* gsrc) {
  unsigned int loff = (unsigned int)(uintptr_t)lds_dst;
  unsigned long long ga = (unsigned long long)(uintptr_t)gsrc;
  asm volatile("global_load_async_to_lds_b128 %0, %1, off"
               :: "v"(loff), "v"(ga)
               : "memory");
}
DEV void wait_async0() { asm volatile("s_wait_asynccnt 0" ::: "memory"); }

// ---------------------------------------------------------------------------
// Generic WMMA GEMM (K fixed at compile time): Y[M,N] = act(X @ WT^T + bias)
// One wave computes a 16x64 strip: A-fragment reused across 4 B tiles;
// K-loop fully unrolled -> straight-line 4*(K/32) WMMA body.
// ACT: 0 none, 1 elu, 2 tanh.  MULAUX: multiply by aux[M,N] (f16) before store.
// grid = (M/16, N/64); M must be a multiple of 16.
// ---------------------------------------------------------------------------
template <int ACT, bool MULAUX, int K>
__global__ __launch_bounds__(32) void gemm_kernel(
    const h16* __restrict__ X, const h16* __restrict__ WT,
    const float* __restrict__ bias, const h16* __restrict__ aux,
    h16* __restrict__ Y, int M, int N) {
  int row0 = blockIdx.x * 16, col0 = blockIdx.y * 64;
  if (row0 + 16 < M) __builtin_prefetch(X + (size_t)(row0 + 16) * K, 0, 1);
  v8f acc[4] = {{}, {}, {}, {}};
#pragma unroll
  for (int k0 = 0; k0 < K; k0 += 32) {
    v16h a = load_a_frag_nc(X, K, row0, k0);
#pragma unroll
    for (int j = 0; j < 4; ++j) {
      v16h b = load_b_frag(WT, K, col0 + j * 16, k0);
      acc[j] = __builtin_amdgcn_wmma_f32_16x16x32_f16(false, a, false, b, (short)0, acc[j], false, false);
    }
  }
  int l = threadIdx.x & 31, hf = l >> 4, n = l & 15;
#pragma unroll
  for (int j = 0; j < 4; ++j) {
    float bv = bias[col0 + j * 16 + n];
#pragma unroll
    for (int r = 0; r < 8; ++r) {
      int row = row0 + r + hf * 8;
      float v = acc[j][r] + bv;
      if (ACT == 1) v = elu1(v);
      if (ACT == 2) v = tanhf(v);
      size_t idx = (size_t)row * N + col0 + j * 16 + n;
      if (MULAUX) v *= (float)aux[idx];
      Y[idx] = (h16)v;
    }
  }
}

// ---------------------------------------------------------------------------
// conv1d (K=3, VALID) as 3 accumulated GEMMs. X:[S,Lin,128] f16, WT3:[3][128][128]
// One wave computes 16 rows x 64 cols. grid = (2 col groups, ceil(Lout/16), S)
// ---------------------------------------------------------------------------
template <int ACT>
__global__ __launch_bounds__(32) void conv3_kernel(
    const h16* __restrict__ X, const h16* __restrict__ WT3,
    const float* __restrict__ bias, h16* __restrict__ Y, int Lin, int Lout) {
  int cg = blockIdx.x, tm = blockIdx.y, seq = blockIdx.z;
  const h16* xs = X + (size_t)seq * Lin * 128;
  h16* ys = Y + (size_t)seq * Lout * 128;
  v8f acc[4] = {{}, {}, {}, {}};
  for (int tap = 0; tap < 3; ++tap) {
    const h16* wt = WT3 + tap * 128 * 128;
#pragma unroll
    for (int k0 = 0; k0 < 128; k0 += 32) {
      v16h a = load_a_frag(xs, 128, tm * 16 + tap, k0, Lin - 1);
#pragma unroll
      for (int j = 0; j < 4; ++j) {
        v16h b = load_b_frag(wt, 128, cg * 64 + j * 16, k0);
        acc[j] = __builtin_amdgcn_wmma_f32_16x16x32_f16(false, a, false, b, (short)0, acc[j], false, false);
      }
    }
  }
  int l = threadIdx.x & 31, hf = l >> 4, n = l & 15;
#pragma unroll
  for (int j = 0; j < 4; ++j) {
    float bv = bias[cg * 64 + j * 16 + n];
#pragma unroll
    for (int r = 0; r < 8; ++r) {
      int row = tm * 16 + r + hf * 8;
      if (row < Lout) {
        float v = acc[j][r] + bv;
        if (ACT == 1) v = elu1(v);
        if (ACT == 2) v = tanhf(v);
        ys[(size_t)row * 128 + cg * 64 + j * 16 + n] = (h16)v;
      }
    }
  }
}

// ---------------------------------------------------------------------------
// Fused self-attention, one block = (sequence, head). 8 waves, wave w owns
// score rows 16w..16w+15.  QKV:[S,L,384] f16 -> O:[S,L,128] f16.
// Q/K staged to LDS via async-to-LDS b128; V staged with transpose via VGPRs.
// ---------------------------------------------------------------------------
__global__ __launch_bounds__(256) void attn_kernel(
    const h16* __restrict__ QKV, h16* __restrict__ O, int L, int ld) {
  __shared__ h16 Qs[128 * 32];
  __shared__ h16 Ks[128 * 32];
  __shared__ h16 Vts[32 * 128];       // V transposed: [n][k]
  __shared__ h16 Ps[8][16 * 128];
  int seq = blockIdx.x, head = blockIdx.y;
  const h16* base = QKV + (size_t)seq * L * ld + head * 32;
  int tid = threadIdx.x;
  // Q,K: async 16B chunks (4 per 32-elem row)
  for (int c = tid; c < L * 4; c += 256) {
    int r = c >> 2, q = c & 3;
    const h16* rowp = base + (size_t)r * ld + q * 8;
    async_copy16(&Qs[r * 32 + q * 8], rowp);
    async_copy16(&Ks[r * 32 + q * 8], rowp + 128);
  }
  // zero-fill padding rows of Q,K
  for (int idx = L * 32 + tid; idx < 128 * 32; idx += 256) {
    Qs[idx] = (h16)0.f; Ks[idx] = (h16)0.f;
  }
  // V with transpose
  for (int idx = tid; idx < 128 * 32; idx += 256) {
    int r = idx >> 5, c = idx & 31;
    h16 v = (h16)0.f;
    if (r < L) v = base[(size_t)r * ld + 256 + c];
    Vts[c * 128 + r] = v;
  }
  wait_async0();
  __syncthreads();

  int wave = tid >> 5, l = tid & 31, hf = l >> 4, nn = l & 15;
  int row0 = wave * 16;
  v16h aq = load_a_frag(Qs, 32, row0, 0, 127);
  v8f sc[8];
#pragma unroll
  for (int j = 0; j < 8; ++j) {
    v16h bk = load_b_frag(Ks, 32, j * 16, 0);
    v8f z = {};
    sc[j] = __builtin_amdgcn_wmma_f32_16x16x32_f16(false, aq, false, bk, (short)0, z, false, false);
  }
  const float scale = 0.17677669529663687f;  // 1/sqrt(32)
  float rsum[8];
#pragma unroll
  for (int r = 0; r < 8; ++r) {
    float m = -1e30f;
#pragma unroll
    for (int j = 0; j < 8; ++j) {
      float v = sc[j][r] * scale; sc[j][r] = v;
      if (j * 16 + nn < L) m = fmaxf(m, v);
    }
#pragma unroll
    for (int x = 1; x < 16; x <<= 1) m = fmaxf(m, __shfl_xor(m, x, 32));
    float s = 0.f;
#pragma unroll
    for (int j = 0; j < 8; ++j) {
      float e = (j * 16 + nn < L) ? __expf(sc[j][r] - m) : 0.f;
      sc[j][r] = e; s += e;
    }
#pragma unroll
    for (int x = 1; x < 16; x <<= 1) s += __shfl_xor(s, x, 32);
    rsum[r] = s;
  }
  h16* pw = &Ps[wave][0];
#pragma unroll
  for (int r = 0; r < 8; ++r) {
    int rl = r + hf * 8;
#pragma unroll
    for (int j = 0; j < 8; ++j) pw[rl * 128 + j * 16 + nn] = (h16)sc[j][r];
  }
  __syncthreads();
  v8f oc0 = {}, oc1 = {};
#pragma unroll
  for (int kb = 0; kb < 128; kb += 32) {
    v16h ap = load_a_frag(pw, 128, 0, kb, 15);
    v16h b0 = load_b_frag(Vts, 128, 0, kb);
    v16h b1 = load_b_frag(Vts, 128, 16, kb);
    oc0 = __builtin_amdgcn_wmma_f32_16x16x32_f16(false, ap, false, b0, (short)0, oc0, false, false);
    oc1 = __builtin_amdgcn_wmma_f32_16x16x32_f16(false, ap, false, b1, (short)0, oc1, false, false);
  }
#pragma unroll
  for (int r = 0; r < 8; ++r) {
    int row = row0 + r + hf * 8;
    if (row < L) {
      float inv = 1.f / rsum[r];
      O[((size_t)seq * L + row) * 128 + head * 32 + nn]      = (h16)(oc0[r] * inv);
      O[((size_t)seq * L + row) * 128 + head * 32 + 16 + nn] = (h16)(oc1[r] * inv);
    }
  }
}

// ---------------------------------------------------------------------------
// Cross-attention with query length 1. KV:[S,L,256] f16 (K cols 0..127, V 128..255)
// grid=S, block=128 (head = wave)
// ---------------------------------------------------------------------------
__global__ __launch_bounds__(128) void cross_attn_kernel(
    const h16* __restrict__ KV, const float* __restrict__ qp,
    h16* __restrict__ O, int L) {
  __shared__ float sc[4][128];
  __shared__ float qs[128];
  int seq = blockIdx.x, tid = threadIdx.x;
  int head = tid >> 5, lane = tid & 31;
  const h16* base = KV + (size_t)seq * L * 256;
  qs[tid] = qp[tid];
  __syncthreads();
  const float scale = 0.17677669529663687f;
  for (int pos = lane; pos < L; pos += 32) {
    float s = 0.f;
    for (int d = 0; d < 32; ++d)
      s += qs[head * 32 + d] * (float)base[(size_t)pos * 256 + head * 32 + d];
    sc[head][pos] = s * scale;
  }
  __syncthreads();
  float m = -1e30f;
  for (int pos = lane; pos < L; pos += 32) m = fmaxf(m, sc[head][pos]);
  for (int x = 1; x < 32; x <<= 1) m = fmaxf(m, __shfl_xor(m, x, 32));
  float ssum = 0.f;
  for (int pos = lane; pos < L; pos += 32) { float e = __expf(sc[head][pos] - m); sc[head][pos] = e; ssum += e; }
  for (int x = 1; x < 32; x <<= 1) ssum += __shfl_xor(ssum, x, 32);
  __syncthreads();
  float o = 0.f;
  for (int pos = 0; pos < L; ++pos)
    o += sc[head][pos] * (float)base[(size_t)pos * 256 + 128 + head * 32 + lane];
  O[(size_t)seq * 128 + head * 32 + lane] = (h16)(o / ssum);
}

// ---------------------------------------------------------------------------
// LayerNorm over cnt elements per sample, in-place f16. grid = nsamples.
// ---------------------------------------------------------------------------
__global__ __launch_bounds__(256) void ln_kernel(h16* __restrict__ X, int cnt) {
  __shared__ float red[256];
  int b = blockIdx.x, tid = threadIdx.x;
  h16* x = X + (size_t)b * cnt;
  float s = 0.f, s2 = 0.f;
  for (int i = tid; i < cnt; i += 256) { float v = (float)x[i]; s += v; s2 += v * v; }
  red[tid] = s; __syncthreads();
  for (int o = 128; o > 0; o >>= 1) { if (tid < o) red[tid] += red[tid + o]; __syncthreads(); }
  float mean = red[0] / (float)cnt; __syncthreads();
  red[tid] = s2; __syncthreads();
  for (int o = 128; o > 0; o >>= 1) { if (tid < o) red[tid] += red[tid + o]; __syncthreads(); }
  float inv = rsqrtf(red[0] / (float)cnt - mean * mean + 1e-5f);
  for (int i = tid; i < cnt; i += 256) x[i] = (h16)(((float)x[i] - mean) * inv);
}

// ---------------------------------------------------------------------------
// Small helper kernels
// ---------------------------------------------------------------------------
__global__ void cast_f16_kernel(const float* __restrict__ s, h16* __restrict__ d, int n) {
  int i = blockIdx.x * 256 + threadIdx.x;
  if (i < n) d[i] = (h16)s[i];
}

// per-tap [O,I] from torch conv weight [O,I,3]
__global__ void pack_conv_kernel(const float* __restrict__ w, h16* __restrict__ dst) {
  int i = blockIdx.x * 256 + threadIdx.x;
  if (i >= 3 * 128 * 128) return;
  int tap = i / 16384, o = (i / 128) & 127, ii = i & 127;
  dst[i] = (h16)w[(o * 128 + ii) * 3 + tap];
}

__global__ void embed_kernel(const int* __restrict__ tok, const float* __restrict__ tab,
                             h16* __restrict__ E, int total, int Wd, int Dd) {
  int i = blockIdx.x * 256 + threadIdx.x;
  if (i >= total) return;
  int d = i % Dd, w = (i / Dd) % Wd, s = i / (Dd * Wd);
  int t = tok[s * Wd + w];
  E[i] = (h16)(tab[t * Dd + d] + pe_val(w, d, Dd));
}

__global__ void add_posenc_kernel(h16* __restrict__ X, int total, int Lp, int Dd) {
  int i = blockIdx.x * 256 + threadIdx.x;
  if (i >= total) return;
  int d = i % Dd, pos = (i / Dd) % Lp;
  X[i] = (h16)((float)X[i] + pe_val(pos, d, Dd));
}

__global__ void qproj_kernel(const float* __restrict__ tgt, const float* __restrict__ wi,
                             const float* __restrict__ bi, float* __restrict__ qp) {
  int o = threadIdx.x;
  float s = bi[o];
  for (int i = 0; i < 128; ++i) s += tgt[i] * wi[o * 128 + i];
  qp[o] = s;
}

__global__ __launch_bounds__(128) void vae_head_kernel(
    const h16* __restrict__ doc, const float* wmu, const float* bmu,
    const float* wlv, const float* blv, const float* __restrict__ eps,
    float* __restrict__ mu, float* __restrict__ lv, float* __restrict__ z) {
  __shared__ float db[128];
  int b = blockIdx.x, d = threadIdx.x;
  db[d] = (float)doc[b * 128 + d];
  __syncthreads();
  float smu = bmu[d], slv = blv[d];
  for (int i = 0; i < 128; ++i) { float x = db[i]; smu += x * wmu[d * 128 + i]; slv += x * wlv[d * 128 + i]; }
  mu[b * 128 + d] = smu;
  lv[b * 128 + d] = slv;
  z[b * 128 + d] = smu + eps[b * 128 + d] * __expf(0.5f * slv);
}

// ---------------------------------------------------------------------------
// Decoder mega-kernel: one block per batch element, 128 threads.
// ---------------------------------------------------------------------------
DEV void block_ln(float* buf, int n, float* red) {
  int tid = threadIdx.x;
  float s = 0.f, s2 = 0.f;
  for (int i = tid; i < n; i += 128) { float v = buf[i]; s += v; s2 += v * v; }
  red[tid] = s; __syncthreads();
  for (int o = 64; o > 0; o >>= 1) { if (tid < o) red[tid] += red[tid + o]; __syncthreads(); }
  float mean = red[0] / (float)n; __syncthreads();
  red[tid] = s2; __syncthreads();
  for (int o = 64; o > 0; o >>= 1) { if (tid < o) red[tid] += red[tid + o]; __syncthreads(); }
  float inv = rsqrtf(red[0] / (float)n - mean * mean + 1e-5f); __syncthreads();
  for (int i = tid; i < n; i += 128) buf[i] = (buf[i] - mean) * inv;
  __syncthreads();
}

__global__ __launch_bounds__(128) void decoder_kernel(
    const float* __restrict__ Z, const float* w_di, const float* b_di,
    const float* dkw, const float* dkb, const float* dvw, const float* dvb,
    const float* wt1, const float* bt1, const float* wt2, const float* bt2,
    const float* sqw, const float* sqb, const float* skw, const float* skb,
    const float* svw, const float* svb, const float* wt3, const float* bt3,
    float* __restrict__ recon) {
  __shared__ float h0[128], kk[128], vv[128], s1[64], h1[128];
  __shared__ float hb1[64 * 7];
  __shared__ float hb2[128 * 13];
  __shared__ float qc[128 * 13], kc2[128 * 13], vc2[128 * 13];
  __shared__ float att2[4 * 13 * 13];
  __shared__ float red[128];
  int b = blockIdx.x, tid = threadIdx.x;
  {
    float s = b_di[tid]; const float* zb = Z + b * 128;
    for (int i = 0; i < 128; ++i) s += zb[i] * w_di[tid * 128 + i];
    h0[tid] = s;
  }
  __syncthreads();
  {
    int o = tid >> 2, l2 = tid & 3;
    float sk = dkb[o], sv = dvb[o];
    for (int i = 0; i < 32; ++i) { float x = h0[i * 4 + l2]; sk += dkw[o * 32 + i] * x; sv += dvw[o * 32 + i] * x; }
    kk[tid] = sk; vv[tid] = sv;
  }
  __syncthreads();
  if (tid < 64) {
    int hh = tid >> 4, l2 = (tid >> 2) & 3, m = tid & 3;
    float s = 0.f;
    for (int d = 0; d < 8; ++d) s += kk[(hh * 8 + d) * 4 + l2] * vv[(hh * 8 + d) * 4 + m];
    s1[tid] = s * 0.35355339059327373f;  // 1/sqrt(8)
  }
  __syncthreads();
  if (tid < 16) {
    int base = tid * 4;
    float m = fmaxf(fmaxf(s1[base], s1[base + 1]), fmaxf(s1[base + 2], s1[base + 3]));
    float e0 = __expf(s1[base] - m), e1 = __expf(s1[base + 1] - m);
    float e2 = __expf(s1[base + 2] - m), e3 = __expf(s1[base + 3] - m);
    float s = e0 + e1 + e2 + e3;
    s1[base] = e0 / s; s1[base + 1] = e1 / s; s1[base + 2] = e2 / s; s1[base + 3] = e3 / s;
  }
  __syncthreads();
  {
    int hh = tid >> 5, rem = tid & 31, l2 = rem >> 3, d = rem & 7;
    float s = 0.f;
    for (int m = 0; m < 4; ++m) s += s1[hh * 16 + l2 * 4 + m] * vv[(hh * 8 + d) * 4 + m];
    h1[hh * 32 + l2 * 8 + d] = elu1(s);
  }
  __syncthreads();
  for (int idx = tid; idx < 64 * 7; idx += 128) {
    int o = idx / 7, t = idx % 7; float s = bt1[o];
    for (int k = 0; k < 3; ++k) {
      int tt = t + 1 - k;
      if (tt >= 0 && !(tt & 1)) { int j = tt >> 1; if (j < 4)
        for (int i = 0; i < 32; ++i) s += h1[i * 4 + j] * wt1[(i * 64 + o) * 3 + k]; }
    }
    hb1[idx] = s;
  }
  __syncthreads();
  block_ln(hb1, 448, red);
  for (int idx = tid; idx < 448; idx += 128) hb1[idx] = elu1(hb1[idx]);
  __syncthreads();
  for (int idx = tid; idx < 128 * 13; idx += 128) {
    int o = idx / 13, t = idx % 13; float s = bt2[o];
    for (int k = 0; k < 3; ++k) {
      int tt = t + 1 - k;
      if (tt >= 0 && !(tt & 1)) { int j = tt >> 1; if (j < 7)
        for (int i = 0; i < 64; ++i) s += hb1[i * 7 + j] * wt2[(i * 128 + o) * 3 + k]; }
    }
    hb2[idx] = s;
  }
  __syncthreads();
  block_ln(hb2, 1664, red);
  for (int idx = tid; idx < 1664; idx += 128) hb2[idx] = elu1(hb2[idx]);
  __syncthreads();
  for (int idx = tid; idx < 1664; idx += 128) {
    int o = idx / 13, l2 = idx % 13;
    float a = sqb[o], c = skb[o], v2 = svb[o];
    for (int i = 0; i < 128; ++i) {
      float x = hb2[i * 13 + l2];
      a += sqw[o * 128 + i] * x; c += skw[o * 128 + i] * x; v2 += svw[o * 128 + i] * x;
    }
    qc[idx] = a; kc2[idx] = c; vc2[idx] = v2;
  }
  __syncthreads();
  for (int idx = tid; idx < 4 * 169; idx += 128) {
    int hh = idx / 169, rem = idx % 169, l2 = rem / 13, m = rem % 13;
    float s = 0.f;
    for (int d = 0; d < 32; ++d) s += qc[(hh * 32 + d) * 13 + l2] * kc2[(hh * 32 + d) * 13 + m];
    att2[idx] = s * 0.17677669529663687f;  // 1/sqrt(32)
  }
  __syncthreads();
  if (tid < 52) {
    int hh = tid / 13, l2 = tid % 13; float* rowp = &att2[hh * 169 + l2 * 13];
    float m = -1e30f;
    for (int j = 0; j < 13; ++j) m = fmaxf(m, rowp[j]);
    float s = 0.f;
    for (int j = 0; j < 13; ++j) { float e = __expf(rowp[j] - m); rowp[j] = e; s += e; }
    float inv = 1.f / s;
    for (int j = 0; j < 13; ++j) rowp[j] *= inv;
  }
  __syncthreads();
  for (int idx = tid; idx < 1664; idx += 128) {
    int hh = idx / 416, rem = idx % 416, l2 = rem / 32, dd = rem & 31;
    float s = 0.f;
    for (int m = 0; m < 13; ++m) s += att2[hh * 169 + l2 * 13 + m] * vc2[hh * 416 + m * 32 + dd];
    hb2[idx] = elu1(s);
  }
  __syncthreads();
  for (int idx = tid; idx < 1664; idx += 128) {
    int o = idx / 13, t = idx % 13; float s = bt3[o];
    for (int k = 0; k < 3; ++k) {
      int j = t + 1 - k;
      if (j >= 0 && j < 13)
        for (int i = 0; i < 128; ++i) s += hb2[i * 13 + j] * wt3[(i * 128 + o) * 3 + k];
    }
    qc[idx] = s;
  }
  __syncthreads();
  block_ln(qc, 1664, red);
  for (int idx = tid; idx < 1664; idx += 128)
    recon[b * 1664 + idx] = 1.f / (1.f + __expf(-qc[idx]));
}

// ---------------------------------------------------------------------------
// Host orchestration
// ---------------------------------------------------------------------------
extern "C" void kernel_launch(void* const* d_in, const int* in_sizes, int n_in,
                              void* d_out, int out_size, void* d_ws, size_t ws_size,
                              hipStream_t stream) {
  (void)in_sizes; (void)n_in; (void)out_size; (void)ws_size;
  const int*   tokens = (const int*)d_in[0];
  const float* table  = (const float*)d_in[1];
  const float* w_ce  = (const float*)d_in[2],  *b_ce  = (const float*)d_in[3];
  const float* w_ct  = (const float*)d_in[4],  *b_ct  = (const float*)d_in[5];
  const float* w_cg  = (const float*)d_in[6],  *b_cg  = (const float*)d_in[7];
  const float* w_ces = (const float*)d_in[8],  *b_ces = (const float*)d_in[9];
  const float* w_cts = (const float*)d_in[10], *b_cts = (const float*)d_in[11];
  const float* wi = (const float*)d_in[12], *bi = (const float*)d_in[13];
  const float* wo = (const float*)d_in[14], *bo = (const float*)d_in[15];
  const float* tgt1 = (const float*)d_in[16], *tgt2 = (const float*)d_in[17];
  const float* w_mu = (const float*)d_in[18], *b_mu = (const float*)d_in[19];
  const float* w_lv = (const float*)d_in[20], *b_lv = (const float*)d_in[21];
  const float* w_di = (const float*)d_in[22], *b_di = (const float*)d_in[23];
  const float* eps = (const float*)d_in[24];
  const float* dk_w = (const float*)d_in[25], *dk_b = (const float*)d_in[26];
  const float* dv_w = (const float*)d_in[27], *dv_b = (const float*)d_in[28];
  const float* wt1 = (const float*)d_in[29], *bt1 = (const float*)d_in[30];
  const float* wt2 = (const float*)d_in[31], *bt2 = (const float*)d_in[32];
  const float* sq_w = (const float*)d_in[33], *sq_b = (const float*)d_in[34];
  const float* sk_w = (const float*)d_in[35], *sk_b = (const float*)d_in[36];
  const float* sv_w = (const float*)d_in[37], *sv_b = (const float*)d_in[38];
  const float* wt3 = (const float*)d_in[39], *bt3 = (const float*)d_in[40];
  float* out = (float*)d_out;

  const int BN = 1024, Wd = 128, Dd = 128, LW = 126, LG = 124, NS = 62;

  char* ws = (char*)d_ws;
  size_t off = 0;
  auto alloc = [&](size_t bytes) -> char* {
    char* p = ws + off;
    off = (off + bytes + 255) & ~(size_t)255;
    return p;
  };
  h16* wi16  = (h16*)alloc((size_t)384 * 128 * 2);
  h16* wo16  = (h16*)alloc((size_t)128 * 128 * 2);
  h16* pk_ce  = (h16*)alloc((size_t)3 * 128 * 128 * 2);
  h16* pk_ct  = (h16*)alloc((size_t)3 * 128 * 128 * 2);
  h16* pk_cg  = (h16*)alloc((size_t)3 * 128 * 128 * 2);
  h16* pk_ces = (h16*)alloc((size_t)3 * 128 * 128 * 2);
  h16* pk_cts = (h16*)alloc((size_t)3 * 128 * 128 * 2);
  float* qp1 = (float*)alloc(128 * 4);
  float* qp2 = (float*)alloc(128 * 4);
  h16* Eb   = (h16*)alloc((size_t)BN * Wd * Dd * 2);      // 33.5 MB
  h16* Hb   = (h16*)alloc((size_t)BN * LW * Dd * 2);      // 33 MB
  h16* QKVb = (h16*)alloc((size_t)BN * LW * 384 * 2);     // 99 MB
  h16* AOb  = (h16*)alloc((size_t)BN * LW * Dd * 2);      // 33 MB
  h16* Gb   = (h16*)alloc((size_t)BN * LW * Dd * 2);      // 33 MB
  h16* SEb  = (h16*)alloc((size_t)16 * 64 * Dd * 2);
  h16* DOCb = (h16*)alloc((size_t)16 * Dd * 2);
  float* Zb = (float*)alloc((size_t)16 * Dd * 4);

  // ---- weight packing ----
  cast_f16_kernel<<<(384 * 128 + 255) / 256, 256, 0, stream>>>(wi, wi16, 384 * 128);
  cast_f16_kernel<<<(128 * 128 + 255) / 256, 256, 0, stream>>>(wo, wo16, 128 * 128);
  const int CP = 3 * 128 * 128;
  pack_conv_kernel<<<(CP + 255) / 256, 256, 0, stream>>>(w_ce, pk_ce);
  pack_conv_kernel<<<(CP + 255) / 256, 256, 0, stream>>>(w_ct, pk_ct);
  pack_conv_kernel<<<(CP + 255) / 256, 256, 0, stream>>>(w_cg, pk_cg);
  pack_conv_kernel<<<(CP + 255) / 256, 256, 0, stream>>>(w_ces, pk_ces);
  pack_conv_kernel<<<(CP + 255) / 256, 256, 0, stream>>>(w_cts, pk_cts);
  qproj_kernel<<<1, 128, 0, stream>>>(tgt1, wi, bi, qp1);
  qproj_kernel<<<1, 128, 0, stream>>>(tgt2, wi, bi, qp2);

  // ---- embedding + positional encoding ----
  int etot = BN * Wd * Dd;
  embed_kernel<<<(etot + 255) / 256, 256, 0, stream>>>(tokens, table, Eb, etot, Wd, Dd);

  // ---- word level, branch e (elu) ----
  conv3_kernel<1><<<dim3(2, 8, BN), 32, 0, stream>>>(Eb, pk_ce, b_ce, Hb, Wd, LW);
  gemm_kernel<0, false, 128><<<dim3(BN * LW / 16, 6), 32, 0, stream>>>(Hb, wi16, bi, nullptr, QKVb, BN * LW, 384);
  attn_kernel<<<dim3(BN, 4), 256, 0, stream>>>(QKVb, AOb, LW, 384);
  gemm_kernel<1, false, 128><<<dim3(BN * LW / 16, 2), 32, 0, stream>>>(AOb, wo16, bo, nullptr, Gb, BN * LW, 128);

  // ---- word level, branch t (tanh); fuse g = a_e * tanh(proj) ----
  conv3_kernel<2><<<dim3(2, 8, BN), 32, 0, stream>>>(Eb, pk_ct, b_ct, Hb, Wd, LW);
  gemm_kernel<0, false, 128><<<dim3(BN * LW / 16, 6), 32, 0, stream>>>(Hb, wi16, bi, nullptr, QKVb, BN * LW, 384);
  attn_kernel<<<dim3(BN, 4), 256, 0, stream>>>(QKVb, AOb, LW, 384);
  gemm_kernel<2, true, 128><<<dim3(BN * LW / 16, 2), 32, 0, stream>>>(AOb, wo16, bo, Gb, Gb, BN * LW, 128);

  // ---- LN per document over N*(W-2)*D ----
  ln_kernel<<<16, 256, 0, stream>>>(Gb, 64 * LW * Dd);

  // ---- conv cg + elu, then KV proj + cross-attn (q1) + out-proj ----
  conv3_kernel<1><<<dim3(2, 8, BN), 32, 0, stream>>>(Gb, pk_cg, b_cg, Hb, LW, LG);
  gemm_kernel<0, false, 128><<<dim3(BN * LG / 16, 4), 32, 0, stream>>>(Hb, wi16 + 128 * 128, bi + 128, nullptr, QKVb, BN * LG, 256);
  cross_attn_kernel<<<BN, 128, 0, stream>>>(QKVb, qp1, AOb, LG);
  gemm_kernel<0, false, 128><<<dim3(BN / 16, 2), 32, 0, stream>>>(AOb, wo16, bo, nullptr, SEb, BN, 128);
  add_posenc_kernel<<<(16 * 64 * Dd + 255) / 256, 256, 0, stream>>>(SEb, 16 * 64 * Dd, 64, Dd);

  // ---- sentence level, branch e ----
  conv3_kernel<1><<<dim3(2, 4, 16), 32, 0, stream>>>(SEb, pk_ces, b_ces, Hb, 64, NS);
  gemm_kernel<0, false, 128><<<dim3(16 * NS / 16, 6), 32, 0, stream>>>(Hb, wi16, bi, nullptr, QKVb, 16 * NS, 384);
  attn_kernel<<<dim3(16, 4), 256, 0, stream>>>(QKVb, AOb, NS, 384);
  gemm_kernel<1, false, 128><<<dim3(16 * NS / 16, 2), 32, 0, stream>>>(AOb, wo16, bo, nullptr, Gb, 16 * NS, 128);
  // ---- sentence level, branch t ----
  conv3_kernel<2><<<dim3(2, 4, 16), 32, 0, stream>>>(SEb, pk_cts, b_cts, Hb, 64, NS);
  gemm_kernel<0, false, 128><<<dim3(16 * NS / 16, 6), 32, 0, stream>>>(Hb, wi16, bi, nullptr, QKVb, 16 * NS, 384);
  attn_kernel<<<dim3(16, 4), 256, 0, stream>>>(QKVb, AOb, NS, 384);
  gemm_kernel<2, true, 128><<<dim3(16 * NS / 16, 2), 32, 0, stream>>>(AOb, wo16, bo, Gb, Gb, 16 * NS, 128);
  ln_kernel<<<16, 256, 0, stream>>>(Gb, NS * Dd);

  // ---- doc cross-attn (q2) + out-proj ----
  gemm_kernel<0, false, 128><<<dim3(16 * NS / 16, 4), 32, 0, stream>>>(Gb, wi16 + 128 * 128, bi + 128, nullptr, QKVb, 16 * NS, 256);
  cross_attn_kernel<<<16, 128, 0, stream>>>(QKVb, qp2, AOb, NS);
  gemm_kernel<0, false, 128><<<dim3(1, 2), 32, 0, stream>>>(AOb, wo16, bo, nullptr, DOCb, 16, 128);

  // ---- VAE head: mu/lv straight to d_out, z to ws ----
  vae_head_kernel<<<16, 128, 0, stream>>>(DOCb, w_mu, b_mu, w_lv, b_lv, eps,
                                          out + 26624, out + 28672, Zb);

  // ---- decoder (tiny): recon to d_out[0:26624] ----
  decoder_kernel<<<16, 128, 0, stream>>>(Zb, w_di, b_di, dk_w, dk_b, dv_w, dv_b,
                                         wt1, bt1, wt2, bt2, sq_w, sq_b, sk_w, sk_b,
                                         sv_w, sv_b, wt3, bt3, out);
}
